// MultiHeadAttention_28140625723752
// MI455X (gfx1250) — matrix operand order repaired and checked by reading, not compiled, not run on hardware
//
#include <hip/hip_runtime.h>
#include <hip/hip_bf16.h>

#define B_  4
#define H_  16
#define S_  2048
#define D_  1024
#define DK_ 64
#define NKT (S_ / 64)

typedef __attribute__((ext_vector_type(16))) _Float16 v16h;
typedef __attribute__((ext_vector_type(8)))  _Float16 v8h;
typedef __attribute__((ext_vector_type(4)))  _Float16 v4h;
typedef __attribute__((ext_vector_type(8)))  float    v8f;

static __device__ __forceinline__ v16h cat16(v8h a, v8h b) {
  return __builtin_shufflevector(a, b, 0,1,2,3,4,5,6,7,8,9,10,11,12,13,14,15);
}
static __device__ __forceinline__ v8f wmma16(v16h a, v16h b, v8f c) {
  // D = A(16x32 f16) * B(32x16 f16) + C(16x16 f32)
  return __builtin_amdgcn_wmma_f32_16x16x32_f16(false, a, false, b, (short)0, c,
                                                false, false);
}

// ---------------------------------------------------------------------------
// proj_kernel: Y[M,N] = X[M,K] @ W[N,K]^T + bias   (M=8192, N=K=1024)
// MODE 0: X is f32, output f16 in split-head layout [B][H][S][DK]
// MODE 1: X is f16, output f32 row-major [M][N]  (final projection)
// Block: 256 threads (8 waves), tile 128(M) x 64(N), K-step 32,
// double-buffered LDS: one barrier per K-step, next tile's global loads
// are issued before the current tile's WMMAs.
// ---------------------------------------------------------------------------
template<int MODE>
__global__ __launch_bounds__(256) void proj_kernel(
    const float* __restrict__ Xf, const _Float16* __restrict__ Xh,
    const float* __restrict__ W,  const float* __restrict__ bias,
    _Float16* __restrict__ out16, float* __restrict__ out32)
{
  __shared__ __align__(16) _Float16 Xs[2][128][40]; // 32 K cols + pad
  __shared__ __align__(16) _Float16 Ws[2][64][40];

  const int tid  = threadIdx.x;
  const int w    = tid >> 5;
  const int lane = tid & 31;
  const int m0   = blockIdx.y << 7;   // 128-row tile
  const int n0   = blockIdx.x << 6;   // 64-col tile

  auto stage = [&](int buf, int k0) {
    if (MODE == 0) {
      #pragma unroll
      for (int i = 0; i < 4; ++i) {
        const int idx = tid + (i << 8);
        const int row = idx >> 3, c4 = (idx & 7) << 2;
        const float4 x = *(const float4*)(Xf + (size_t)(m0 + row) * D_ + k0 + c4);
        v4h hx = {(_Float16)x.x, (_Float16)x.y, (_Float16)x.z, (_Float16)x.w};
        *(v4h*)&Xs[buf][row][c4] = hx;
      }
    } else {
      #pragma unroll
      for (int i = 0; i < 2; ++i) {
        const int idx = tid + (i << 8);
        const int row = idx >> 2, c8 = (idx & 3) << 3;
        *(v8h*)&Xs[buf][row][c8] =
            *(const v8h*)(Xh + (size_t)(m0 + row) * D_ + k0 + c8);
      }
    }
    #pragma unroll
    for (int i = 0; i < 2; ++i) {
      const int idx = tid + (i << 8);
      const int row = idx >> 3, c4 = (idx & 7) << 2;
      const float4 x = *(const float4*)(W + (size_t)(n0 + row) * D_ + k0 + c4);
      v4h hx = {(_Float16)x.x, (_Float16)x.y, (_Float16)x.z, (_Float16)x.w};
      *(v4h*)&Ws[buf][row][c4] = hx;
    }
  };

  stage(0, 0);
  __syncthreads();

  v8f acc[4] = {};
  const int rA  = (w << 4) + (lane & 15);
  const int koA = (lane >> 4) << 3;
  const int k16 = (lane >> 4) << 4;

  for (int step = 0; step < D_ / 32; ++step) {
    const int cur = step & 1;
    if (step + 1 < D_ / 32) stage(cur ^ 1, (step + 1) << 5);

    const v16h A = cat16(*(const v8h*)&Xs[cur][rA][koA],
                         *(const v8h*)&Xs[cur][rA][16 + koA]);
    #pragma unroll
    for (int nt = 0; nt < 4; ++nt) {
      const int n = (nt << 4) + (lane & 15);
      const v16h Bf = cat16(*(const v8h*)&Ws[cur][n][koA],
                            *(const v8h*)&Ws[cur][n][16 + koA]);
      acc[nt] = wmma16(A, Bf, acc[nt]);
    }
    __syncthreads();
  }

  // epilogue: C layout -> (m = 16w + 8*(lane>=16) + i, n = lane&15)
  const int lm = (lane >> 4) << 3;
  #pragma unroll
  for (int nt = 0; nt < 4; ++nt) {
    const int ng = n0 + (nt << 4) + (lane & 15);
    const float bv = bias[ng];
    #pragma unroll
    for (int i = 0; i < 8; ++i) {
      const int mg = m0 + (w << 4) + lm + i;
      const float y = acc[nt][i] + bv;
      if (MODE == 0) {
        const int bb = mg >> 11, ss = mg & (S_ - 1);
        const int hh = ng >> 6,  dk = ng & 63;
        out16[(((size_t)(bb * H_ + hh) * S_ + ss) << 6) + dk] = (_Float16)y;
      } else {
        out32[(size_t)mg * D_ + ng] = y;
      }
    }
  }
}

// ---------------------------------------------------------------------------
// attn_kernel: flash-style attention for one (b,h) and a 64-query tile.
// Block: 128 threads (4 waves); each wave owns a 16-row query strip.
// Double-buffered K/V staging: global loads for key-tile kt+1 are issued
// before the QK^T / softmax / PV work of tile kt.
// ---------------------------------------------------------------------------
__global__ __launch_bounds__(128) void attn_kernel(
    const _Float16* __restrict__ qh, const _Float16* __restrict__ kh,
    const _Float16* __restrict__ vh, const int* __restrict__ mask,
    _Float16* __restrict__ aws)
{
  __shared__ __align__(16) _Float16 Ks[2][64][72];  // [key][dk]
  __shared__ __align__(16) _Float16 Vt[2][64][72];  // [dk][key] (V transposed)
  __shared__ __align__(16) float    Sc[64][68];     // scores / probs
  __shared__ float rowm[64], rowl[64], rfac[64];

  const int tid  = threadIdx.x;
  const int w    = tid >> 5;
  const int lane = tid & 31;
  const int q0   = blockIdx.x << 6;
  const int bh   = blockIdx.y;          // b*H + h
  const int b    = bh >> 4, h = bh & 15;
  const size_t bhS = (size_t)bh * S_;

  auto stageKV = [&](int buf, int kt) {
    const _Float16* kbase = kh + ((bhS + ((size_t)kt << 6)) << 6);
    const _Float16* vbase = vh + ((bhS + ((size_t)kt << 6)) << 6);
    #pragma unroll
    for (int i = 0; i < 4; ++i) {
      const int idx = tid + (i << 7);
      const int row = idx >> 3, c8 = (idx & 7) << 3;
      *(v8h*)&Ks[buf][row][c8] = *(const v8h*)(kbase + (row << 6) + c8);
      const v8h vv = *(const v8h*)(vbase + (row << 6) + c8);
      #pragma unroll
      for (int j = 0; j < 8; ++j) Vt[buf][c8 + j][row] = vv[j];
    }
  };

  if (tid < 64) { rowm[tid] = -1e30f; rowl[tid] = 0.0f; }

  // Q fragments held in registers for the whole block (K dim = 64 -> 2 frags)
  const int rA  = lane & 15;
  const int koA = (lane >> 4) << 3;
  const int k16 = (lane >> 4) << 4;
  const int mlb = (w << 4) + ((lane >> 4) << 3);
  const _Float16* qrow = qh + ((bhS + q0 + (w << 4) + rA) << 6);
  const v16h AQ0 = cat16(*(const v8h*)(qrow + koA),      *(const v8h*)(qrow + 16 + koA));
  const v16h AQ1 = cat16(*(const v8h*)(qrow + 32 + koA), *(const v8h*)(qrow + 48 + koA));

  stageKV(0, 0);
  v8f accO[4] = {};

  for (int kt = 0; kt < NKT; ++kt) {
    const int cur = kt & 1;
    __syncthreads();   // staging of `cur` done; prev tile's Sc/Vt reads done
    if (kt + 1 < NKT) stageKV(cur ^ 1, kt + 1);

    // S strip = Q @ K^T  (16 queries x 64 keys per wave)
    #pragma unroll
    for (int nt = 0; nt < 4; ++nt) {
      const int n = (nt << 4) + (lane & 15);
      const v16h B0 = cat16(*(const v8h*)&Ks[cur][n][koA],
                            *(const v8h*)&Ks[cur][n][16 + koA]);
      const v16h B1 = cat16(*(const v8h*)&Ks[cur][n][32 + koA],
                            *(const v8h*)&Ks[cur][n][48 + koA]);
      v8f c = {};
      c = wmma16(AQ0, B0, c);
      c = wmma16(AQ1, B1, c);
      #pragma unroll
      for (int i = 0; i < 8; ++i) Sc[mlb + i][(nt << 4) + (lane & 15)] = c[i];
    }
    __syncthreads();

    // masked online softmax (one thread per query row, vectorized x4)
    if (tid < 64) {
      const int* mrow = mask + ((size_t)b * S_ + q0 + tid) * S_ + (kt << 6);
      if (kt + 1 < NKT) __builtin_prefetch(mrow + 64, 0, 1);
      const float om = rowm[tid];
      float tm = om;
      float4* srow = (float4*)&Sc[tid][0];
      const int4* m4 = (const int4*)mrow;
      #pragma unroll
      for (int j = 0; j < 16; ++j) {
        const int4 mm = m4[j];
        float4 s = srow[j];
        s.x = (mm.x == 0) ? -1e9f : s.x * 0.125f;   // 0.125 = 1/sqrt(dk)
        s.y = (mm.y == 0) ? -1e9f : s.y * 0.125f;
        s.z = (mm.z == 0) ? -1e9f : s.z * 0.125f;
        s.w = (mm.w == 0) ? -1e9f : s.w * 0.125f;
        srow[j] = s;
        tm = fmaxf(tm, fmaxf(fmaxf(s.x, s.y), fmaxf(s.z, s.w)));
      }
      const float fac = __expf(om - tm);
      float l = rowl[tid] * fac;
      #pragma unroll
      for (int j = 0; j < 16; ++j) {
        float4 s = srow[j];
        s.x = __expf(s.x - tm);
        s.y = __expf(s.y - tm);
        s.z = __expf(s.z - tm);
        s.w = __expf(s.w - tm);
        srow[j] = s;
        l += (s.x + s.y) + (s.z + s.w);
      }
      rowm[tid] = tm; rowl[tid] = l; rfac[tid] = fac;
    }
    __syncthreads();

    // rescale running O, then O += P @ V
    float f[8];
    #pragma unroll
    for (int i = 0; i < 8; ++i) f[i] = rfac[mlb + i];
    #pragma unroll
    for (int dt = 0; dt < 4; ++dt)
      #pragma unroll
      for (int i = 0; i < 8; ++i) accO[dt][i] *= f[i];

    const int mr = (w << 4) + rA;
    v16h P0, P1;
    #pragma unroll
    for (int j = 0; j < 8; ++j) {
      P0[j]     = (_Float16)Sc[mr][koA + j];
      P0[8 + j] = (_Float16)Sc[mr][16 + koA + j];
      P1[j]     = (_Float16)Sc[mr][32 + koA + j];
      P1[8 + j] = (_Float16)Sc[mr][48 + koA + j];
    }
    #pragma unroll
    for (int dt = 0; dt < 4; ++dt) {
      const int d = (dt << 4) + (lane & 15);
      const v16h B0 = cat16(*(const v8h*)&Vt[cur][d][koA],
                            *(const v8h*)&Vt[cur][d][16 + koA]);
      const v16h B1 = cat16(*(const v8h*)&Vt[cur][d][32 + koA],
                            *(const v8h*)&Vt[cur][d][48 + koA]);
      accO[dt] = wmma16(P0, B0, accO[dt]);
      accO[dt] = wmma16(P1, B1, accO[dt]);
    }
  }

  // normalize and write f16 in [B][S][H*DK] layout (ready for Wo projection)
  float linv[8];
  #pragma unroll
  for (int i = 0; i < 8; ++i) linv[i] = 1.0f / rowl[mlb + i];
  #pragma unroll
  for (int dt = 0; dt < 4; ++dt) {
    const int dglob = (dt << 4) + (lane & 15);
    #pragma unroll
    for (int i = 0; i < 8; ++i) {
      const int srow = q0 + mlb + i;
      aws[(((size_t)b * S_ + srow) * H_ + h) * DK_ + dglob] =
          (_Float16)(accO[dt][i] * linv[i]);
    }
  }
}

// ---------------------------------------------------------------------------
extern "C" void kernel_launch(void* const* d_in, const int* in_sizes, int n_in,
                              void* d_out, int out_size, void* d_ws, size_t ws_size,
                              hipStream_t stream) {
  const float* q    = (const float*)d_in[0];
  const float* k    = (const float*)d_in[1];
  const float* v    = (const float*)d_in[2];
  const int*   mask = (const int*)d_in[3];
  const float* Wq   = (const float*)d_in[4];
  const float* bq   = (const float*)d_in[5];
  const float* Wk   = (const float*)d_in[6];
  const float* bk   = (const float*)d_in[7];
  const float* Wv   = (const float*)d_in[8];
  const float* bv   = (const float*)d_in[9];
  const float* Wo   = (const float*)d_in[10];
  const float* bo   = (const float*)d_in[11];

  const size_t QKV = (size_t)B_ * S_ * D_;   // 8,388,608 f16 elements
  _Float16* qh  = (_Float16*)d_ws;
  _Float16* kh  = qh + QKV;
  _Float16* vh  = kh + QKV;
  _Float16* aws = vh + QKV;

  dim3 gp(D_ / 64, (B_ * S_) / 128);         // (16, 64)
  proj_kernel<0><<<gp, 256, 0, stream>>>(q, nullptr, Wq, bq, qh, nullptr);
  proj_kernel<0><<<gp, 256, 0, stream>>>(k, nullptr, Wk, bk, kh, nullptr);
  proj_kernel<0><<<gp, 256, 0, stream>>>(v, nullptr, Wv, bv, vh, nullptr);

  dim3 ga(S_ / 64, B_ * H_);                 // (32, 64)
  attn_kernel<<<ga, 128, 0, stream>>>(qh, kh, vh, mask, aws);

  proj_kernel<1><<<gp, 256, 0, stream>>>(nullptr, aws, Wo, bo, nullptr, (float*)d_out);
}
